// GIN_32392643346832
// MI455X (gfx1250) — matrix operand order repaired and checked by reading, not compiled, not run on hardware
//
#include <hip/hip_runtime.h>
#include <hip/hip_bf16.h>
#include <math.h>

#define N_NODES 100000
#define N_EDGES 1600000
#define HID     128
#define NGRAPH  512
#define NCLS    10

typedef __attribute__((ext_vector_type(16))) __bf16 v16bf;
typedef __attribute__((ext_vector_type(8)))  __bf16 v8bf;
typedef __attribute__((ext_vector_type(8)))  float  v8f;

// ---------------------------------------------------------------------------
// Fused GIN MLP: out = relu( relu((hin+agg) @ Wa + ba) @ Wb + bb )
// One block = 64 rows x 128 cols, 8 waves, bf16 WMMA with f32 accumulate.
// LDS: 32KB weight (bf16, transposed, reused for Wa then Wb) + 16KB A-tile.
// ---------------------------------------------------------------------------
__global__ __launch_bounds__(256)
void gin_mlp_kernel(const float* __restrict__ hin, const float* __restrict__ agg,
                    const float* __restrict__ Wa,  const float* __restrict__ ba,
                    const float* __restrict__ Wb,  const float* __restrict__ bb,
                    float* __restrict__ out, int nRows)
{
    __shared__ __align__(16) __bf16 sW[HID * HID];  // sW[n*128 + k] = W[k][n]
    __shared__ __align__(16) __bf16 sX[64 * HID];   // A rows (z, then Y1)

    const int tid     = threadIdx.x;
    const int lane    = tid & 31;
    const int wave    = tid >> 5;
    const int rowBase = blockIdx.x * 64;

    // --- stage z = hin + agg (bf16), zero-pad rows beyond nRows ------------
    for (int q = tid; q < 64 * HID / 4; q += 256) {
        int r  = q >> 5;            // 32 float4 per 128-wide row
        int c4 = (q & 31) * 4;
        int gr = rowBase + r;
        float4 hv = {0.f, 0.f, 0.f, 0.f}, av = {0.f, 0.f, 0.f, 0.f};
        if (gr < nRows) {
            hv = *(const float4*)&hin[(size_t)gr * HID + c4];
            av = *(const float4*)&agg[(size_t)gr * HID + c4];
        }
        __bf16* p = &sX[r * HID + c4];
        p[0] = (__bf16)(hv.x + av.x);
        p[1] = (__bf16)(hv.y + av.y);
        p[2] = (__bf16)(hv.z + av.z);
        p[3] = (__bf16)(hv.w + av.w);
    }
    // --- stage Wa transposed (bf16) ----------------------------------------
    for (int q = tid; q < HID * HID / 4; q += 256) {
        int k  = q >> 5;
        int n4 = (q & 31) * 4;
        float4 w = *(const float4*)&Wa[k * HID + n4];
        sW[(n4 + 0) * HID + k] = (__bf16)w.x;
        sW[(n4 + 1) * HID + k] = (__bf16)w.y;
        sW[(n4 + 2) * HID + k] = (__bf16)w.z;
        sW[(n4 + 3) * HID + k] = (__bf16)w.w;
    }
    __syncthreads();

    const int mt = wave & 3;    // 16-row strip within the 64-row tile
    const int ng = wave >> 2;   // 64-col group (two groups of 4 n-tiles)
    const int hs = lane >> 4;   // half-wave select for K layout
    const int lr = lane & 15;

    v8f acc[4];

    // 64x128 GEMM from sX (A, 16-bit layout) x sW (B, column-major) ---------
    auto matmul = [&]() {
        #pragma unroll
        for (int t = 0; t < 4; ++t)
            #pragma unroll
            for (int j = 0; j < 8; ++j) acc[t][j] = 0.f;
        #pragma unroll
        for (int kk = 0; kk < 4; ++kk) {
            const int kB = kk * 32;
            const int ar = mt * 16 + lr;
            // A frag: lanes 0-15 -> K {0..7,16..23}; lanes 16-31 -> K {8..15,24..31}
            v8bf lo = *(const v8bf*)&sX[ar * HID + kB + hs * 8];
            v8bf hi = *(const v8bf*)&sX[ar * HID + kB + 16 + hs * 8];
            v16bf A;
            #pragma unroll
            for (int j = 0; j < 8; ++j) { A[j] = lo[j]; A[8 + j] = hi[j]; }
            #pragma unroll
            for (int t = 0; t < 4; ++t) {
                const int cb = ng * 64 + t * 16 + lr;
                v8bf blo = *(const v8bf*)&sW[cb * HID + kB + hs * 8];
                v8bf bhi = *(const v8bf*)&sW[cb * HID + kB + 16 + hs * 8];
                v16bf B;
                #pragma unroll
                for (int j = 0; j < 8; ++j) { B[j] = blo[j]; B[8 + j] = bhi[j]; }
                acc[t] = __builtin_amdgcn_wmma_f32_16x16x32_bf16(
                    false, A, false, B, (short)0, acc[t], false, false);
            }
        }
    };

    // ---- layer 1: Y = relu(z @ Wa + ba) -----------------------------------
    matmul();
    __syncthreads();   // all A/W reads done before overwrite

    #pragma unroll
    for (int t = 0; t < 4; ++t) {
        const int cb = ng * 64 + t * 16 + lr;
        const float bias = ba[cb];
        #pragma unroll
        for (int v = 0; v < 8; ++v) {
            const int r = mt * 16 + hs * 8 + v;   // D layout: M = v + 8*hs
            sX[r * HID + cb] = (__bf16)fmaxf(acc[t][v] + bias, 0.f);
        }
    }
    for (int q = tid; q < HID * HID / 4; q += 256) {
        int k  = q >> 5;
        int n4 = (q & 31) * 4;
        float4 w = *(const float4*)&Wb[k * HID + n4];
        sW[(n4 + 0) * HID + k] = (__bf16)w.x;
        sW[(n4 + 1) * HID + k] = (__bf16)w.y;
        sW[(n4 + 2) * HID + k] = (__bf16)w.z;
        sW[(n4 + 3) * HID + k] = (__bf16)w.w;
    }
    __syncthreads();

    // ---- layer 2: out = relu(Y @ Wb + bb) ---------------------------------
    matmul();

    #pragma unroll
    for (int t = 0; t < 4; ++t) {
        const int cb = ng * 64 + t * 16 + lr;
        const float bias = bb[cb];
        #pragma unroll
        for (int v = 0; v < 8; ++v) {
            const int gr = rowBase + mt * 16 + hs * 8 + v;
            if (gr < nRows)
                out[(size_t)gr * HID + cb] = fmaxf(acc[t][v] + bias, 0.f);
        }
    }
}

// ---------------------------------------------------------------------------
// agg[dst] += h[src] over edges. One wave per edge, float4 per lane.
// ---------------------------------------------------------------------------
__global__ __launch_bounds__(256)
void scatter_add_kernel(const float* __restrict__ h, const int* __restrict__ src,
                        const int* __restrict__ dst, float* __restrict__ agg, int nE)
{
    int gid  = blockIdx.x * 256 + threadIdx.x;
    int e    = gid >> 5;
    int lane = gid & 31;
    if (e >= nE) return;
    int s = src[e], d = dst[e];
    float4 v = *(const float4*)&h[(size_t)s * HID + lane * 4];
    float* ap = &agg[(size_t)d * HID + lane * 4];
    atomicAdd(ap + 0, v.x);
    atomicAdd(ap + 1, v.y);
    atomicAdd(ap + 2, v.z);
    atomicAdd(ap + 3, v.w);
}

// ---------------------------------------------------------------------------
// Per-node pooling: atomic sum + atomic max (uint trick, valid for h >= 0).
// ---------------------------------------------------------------------------
__global__ __launch_bounds__(256)
void pool_kernel(const float* __restrict__ h, const int* __restrict__ batch,
                 float* __restrict__ gsum, unsigned int* __restrict__ gmax, int nN)
{
    int gid  = blockIdx.x * 256 + threadIdx.x;
    int n    = gid >> 5;
    int lane = gid & 31;
    if (n >= nN) return;
    int g = batch[n];
    float4 v = *(const float4*)&h[(size_t)n * HID + lane * 4];
    float* sp = &gsum[(size_t)g * HID + lane * 4];
    unsigned int* mp = &gmax[(size_t)g * HID + lane * 4];
    atomicAdd(sp + 0, v.x); atomicMax(mp + 0, __float_as_uint(v.x));
    atomicAdd(sp + 1, v.y); atomicMax(mp + 1, __float_as_uint(v.y));
    atomicAdd(sp + 2, v.z); atomicMax(mp + 2, __float_as_uint(v.z));
    atomicAdd(sp + 3, v.w); atomicMax(mp + 3, __float_as_uint(v.w));
}

__global__ __launch_bounds__(256)
void combine_kernel(const float* __restrict__ gsum, const unsigned int* __restrict__ gmax,
                    const float* __restrict__ cnt, float* __restrict__ gacc)
{
    int idx = blockIdx.x * 256 + threadIdx.x;
    if (idx >= NGRAPH * HID) return;
    int g = idx >> 7, f = idx & 127;
    float s = gsum[idx];
    float c = cnt[g];
    float m = (c > 0.f) ? __uint_as_float(gmax[idx]) : 0.f;
    float* gp = &gacc[(size_t)g * 3 * HID];
    gp[f]           += s / fmaxf(c, 1.f);   // mean
    gp[HID + f]     += m;                   // max
    gp[2 * HID + f] += s;                   // sum
}

__global__ __launch_bounds__(256)
void count_kernel(const int* __restrict__ batch, float* __restrict__ cnt, int nN)
{
    int i = blockIdx.x * 256 + threadIdx.x;
    if (i < nN) atomicAdd(&cnt[batch[i]], 1.f);
}

// ---------------------------------------------------------------------------
// Head: z = relu(g @ Wl1 + bl1)   [512x384 @ 384x128]  (tiny, plain VALU)
// ---------------------------------------------------------------------------
__global__ __launch_bounds__(256)
void head1_kernel(const float* __restrict__ g, const float* __restrict__ W,
                  const float* __restrict__ b, float* __restrict__ z)
{
    int idx = blockIdx.x * 256 + threadIdx.x;
    if (idx >= NGRAPH * HID) return;
    int m = idx >> 7, n = idx & 127;
    float acc = b[n];
    const float* gr = &g[(size_t)m * 3 * HID];
    for (int k = 0; k < 3 * HID; ++k) acc = fmaf(gr[k], W[k * HID + n], acc);
    z[idx] = fmaxf(acc, 0.f);
}

// logits + log_softmax, one wave per graph
__global__ __launch_bounds__(256)
void head2_kernel(const float* __restrict__ z, const float* __restrict__ W,
                  const float* __restrict__ b, float* __restrict__ out)
{
    int wave = threadIdx.x >> 5, lane = threadIdx.x & 31;
    int m = blockIdx.x * 8 + wave;
    if (m >= NGRAPH) return;
    float lg[NCLS];
    #pragma unroll
    for (int c = 0; c < NCLS; ++c) {
        float p = 0.f;
        for (int k = lane; k < HID; k += 32)
            p = fmaf(z[(size_t)m * HID + k], W[k * NCLS + c], p);
        #pragma unroll
        for (int off = 16; off; off >>= 1) p += __shfl_xor(p, off, 32);
        lg[c] = p + b[c];
    }
    if (lane == 0) {
        float mx = lg[0];
        #pragma unroll
        for (int c = 1; c < NCLS; ++c) mx = fmaxf(mx, lg[c]);
        float se = 0.f;
        #pragma unroll
        for (int c = 0; c < NCLS; ++c) se += __expf(lg[c] - mx);
        float lse = mx + __logf(se);
        #pragma unroll
        for (int c = 0; c < NCLS; ++c) out[(size_t)m * NCLS + c] = lg[c] - lse;
    }
}

// ---------------------------------------------------------------------------
extern "C" void kernel_launch(void* const* d_in, const int* in_sizes, int n_in,
                              void* d_out, int out_size, void* d_ws, size_t ws_size,
                              hipStream_t stream)
{
    const float* x    = (const float*)d_in[0];
    const int*   eidx = (const int*)  d_in[1];
    const int*   batch= (const int*)  d_in[2];
    const float* W1a  = (const float*)d_in[3];
    const float* b1a  = (const float*)d_in[4];
    const float* W1b  = (const float*)d_in[5];
    const float* b1b  = (const float*)d_in[6];
    const float* W2a  = (const float*)d_in[7];
    const float* b2a  = (const float*)d_in[8];
    const float* W2b  = (const float*)d_in[9];
    const float* b2b  = (const float*)d_in[10];
    const float* Wl1  = (const float*)d_in[11];
    const float* bl1  = (const float*)d_in[12];
    const float* Wl2  = (const float*)d_in[13];
    const float* bl2  = (const float*)d_in[14];
    const int* src = eidx;
    const int* dst = eidx + N_EDGES;

    char* ws = (char*)d_ws;
    size_t off = 0;
    auto alloc = [&](size_t bytes) {
        char* p = ws + off;
        off += (bytes + 255) & ~size_t(255);
        return p;
    };
    float*        aggb = (float*)alloc((size_t)N_NODES * HID * 4);
    float*        hA   = (float*)alloc((size_t)N_NODES * HID * 4);
    float*        hB   = (float*)alloc((size_t)N_NODES * HID * 4);
    float*        gsum = (float*)alloc((size_t)NGRAPH * HID * 4);
    unsigned int* gmax = (unsigned int*)alloc((size_t)NGRAPH * HID * 4);
    float*        gacc = (float*)alloc((size_t)NGRAPH * 3 * HID * 4);
    float*        cnt  = (float*)alloc((size_t)NGRAPH * 4);
    float*        zh   = (float*)alloc((size_t)NGRAPH * HID * 4);

    const dim3 b256(256);
    const int mlpBlocks  = (N_NODES + 63) / 64;
    const int scatBlocks = (N_EDGES * 32 + 255) / 256;
    const int poolBlocks = (N_NODES * 32 + 255) / 256;

    hipMemsetAsync(cnt, 0, (size_t)NGRAPH * 4, stream);
    hipMemsetAsync(gacc, 0, (size_t)NGRAPH * 3 * HID * 4, stream);
    count_kernel<<<(N_NODES + 255) / 256, b256, 0, stream>>>(batch, cnt, N_NODES);

    auto conv = [&](const float* hin, const float* Wa, const float* ba,
                    const float* Wb, const float* bb, float* hout) {
        hipMemsetAsync(aggb, 0, (size_t)N_NODES * HID * 4, stream);
        scatter_add_kernel<<<scatBlocks, b256, 0, stream>>>(hin, src, dst, aggb, N_EDGES);
        gin_mlp_kernel<<<mlpBlocks, b256, 0, stream>>>(hin, aggb, Wa, ba, Wb, bb, hout, N_NODES);
    };
    auto readout = [&](const float* h) {
        hipMemsetAsync(gsum, 0, (size_t)NGRAPH * HID * 4, stream);
        hipMemsetAsync(gmax, 0, (size_t)NGRAPH * HID * 4, stream);
        pool_kernel<<<poolBlocks, b256, 0, stream>>>(h, batch, gsum, gmax, N_NODES);
        combine_kernel<<<(NGRAPH * HID + 255) / 256, b256, 0, stream>>>(gsum, gmax, cnt, gacc);
    };

    conv(x,  W1a, b1a, W1b, b1b, hA);   // h = relu(conv1(x))
    conv(hA, W2a, b2a, W2b, b2b, hB);   // h = relu(conv2(h))
    readout(hB);
    conv(hB, W2a, b2a, W2b, b2b, hA);   // h = relu(conv3(h))
    readout(hA);
    conv(hA, W2a, b2a, W2b, b2b, hB);   // h = relu(conv4(h))
    readout(hB);

    head1_kernel<<<(NGRAPH * HID + 255) / 256, b256, 0, stream>>>(gacc, Wl1, bl1, zh);
    head2_kernel<<<(NGRAPH + 7) / 8, b256, 0, stream>>>(zh, Wl2, bl2, (float*)d_out);
}
